// RWKVTimeFirst_37864431681708
// MI455X (gfx1250) — matrix operand order repaired and checked by reading, not compiled, not run on hardware
//
#include <hip/hip_runtime.h>
#include <hip/hip_bf16.h>

typedef __attribute__((ext_vector_type(16))) _Float16 v16h;
typedef __attribute__((ext_vector_type(8)))  _Float16 v8h;
typedef __attribute__((ext_vector_type(8)))  float    v8f;
typedef int v4i __attribute__((vector_size(16)));
typedef _Float16 half_t;

#define AS1 __attribute__((address_space(1)))
#define AS3 __attribute__((address_space(3)))

#define B_   4
#define T_   4096
#define C_   1024
#define M_TOT (B_ * T_)          // 16384 rows in every GEMM

// ---- GEMM tiling ----
#define BM 128
#define BN 128
#define BK 32
#define LDS_PAD 8
#define LDS_STRIDE (BK + LDS_PAD)    // halves (40)

// gfx1250 async global->LDS path (ASYNCcnt-tracked), guarded so a missing
// builtin falls back to the synchronous VGPR-bounce staging.
#if defined(__has_builtin)
#  if __has_builtin(__builtin_amdgcn_global_load_async_to_lds_b128) && \
      __has_builtin(__builtin_amdgcn_s_wait_asynccnt)
#    define HAVE_ASYNC_LDS 1
#  endif
#endif
#ifndef HAVE_ASYNC_LDS
#  define HAVE_ASYNC_LDS 0
#endif

// ---------------------------------------------------------------------------
// Tiled WMMA GEMM:  D[m,n] = epi( sum_k A16[m,k] * W16[n,k] )
// A16 : (M x C) row-major fp16 (activation)
// W16 : (C x C) row-major fp16 (weight; row n is exactly B-matrix column n)
// D   : (M x C) row-major fp32
// EPI: 0 = identity, 1 = sigmoid (fast rcp), 2 = exp
// Block: 256 threads = 8 waves, wave grid 4(M) x 2(N), wave tile 32x64:
// 8x v_wmma_f32_16x16x32_f16 per wave per K-step, double-buffered LDS.
// ---------------------------------------------------------------------------
template <int EPI>
__global__ __launch_bounds__(256)
void wmma_gemm_kernel(const half_t* __restrict__ A,
                      const half_t* __restrict__ W,
                      float* __restrict__ D)
{
    __shared__ half_t Alds[2][BM * LDS_STRIDE];
    __shared__ half_t Blds[2][BN * LDS_STRIDE];

    const int tid   = threadIdx.x;
    const int lane  = tid & 31;
    const int wave  = tid >> 5;
    const int waveM = wave >> 1;      // 0..3  (32 rows each)
    const int waveN = wave & 1;       // 0..1  (64 cols each)
    const int lhalf = lane >> 4;      // 0 | 1 (lane half, per WMMA layout)
    const int l16   = lane & 15;

    const int bm = blockIdx.x * BM;   // row tile (over M = B*T)
    const int bn = blockIdx.y * BN;   // col tile (over N = C)

    // per-thread staging coordinates: 512 16B chunks per tile, 2 per thread
    const int arow0 = (tid * 2 + 0) >> 2, acc0 = ((tid * 2 + 0) & 3) * 8;
    const int arow1 = (tid * 2 + 1) >> 2, acc1 = ((tid * 2 + 1) & 3) * 8;

    auto stage = [&](int k0, int buf) {
#if HAVE_ASYNC_LDS
        __builtin_amdgcn_global_load_async_to_lds_b128(
            (AS1 v4i*)(A + (size_t)(bm + arow0) * C_ + k0 + acc0),
            (AS3 v4i*)&Alds[buf][arow0 * LDS_STRIDE + acc0], 0, 0);
        __builtin_amdgcn_global_load_async_to_lds_b128(
            (AS1 v4i*)(A + (size_t)(bm + arow1) * C_ + k0 + acc1),
            (AS3 v4i*)&Alds[buf][arow1 * LDS_STRIDE + acc1], 0, 0);
        __builtin_amdgcn_global_load_async_to_lds_b128(
            (AS1 v4i*)(W + (size_t)(bn + arow0) * C_ + k0 + acc0),
            (AS3 v4i*)&Blds[buf][arow0 * LDS_STRIDE + acc0], 0, 0);
        __builtin_amdgcn_global_load_async_to_lds_b128(
            (AS1 v4i*)(W + (size_t)(bn + arow1) * C_ + k0 + acc1),
            (AS3 v4i*)&Blds[buf][arow1 * LDS_STRIDE + acc1], 0, 0);
#else
        *(v8h*)&Alds[buf][arow0 * LDS_STRIDE + acc0] =
            *(const v8h*)(A + (size_t)(bm + arow0) * C_ + k0 + acc0);
        *(v8h*)&Alds[buf][arow1 * LDS_STRIDE + acc1] =
            *(const v8h*)(A + (size_t)(bm + arow1) * C_ + k0 + acc1);
        *(v8h*)&Blds[buf][arow0 * LDS_STRIDE + acc0] =
            *(const v8h*)(W + (size_t)(bn + arow0) * C_ + k0 + acc0);
        *(v8h*)&Blds[buf][arow1 * LDS_STRIDE + acc1] =
            *(const v8h*)(W + (size_t)(bn + arow1) * C_ + k0 + acc1);
#endif
    };

    v8f acc[2][4] = {};

    stage(0, 0);
#if HAVE_ASYNC_LDS
    __builtin_amdgcn_s_wait_asynccnt(0);
#endif
    __syncthreads();

    const int NSTEP = C_ / BK;        // 32
    for (int it = 0; it < NSTEP; ++it) {
        const int cur = it & 1;
        const int k1 = (it + 1) * BK;
        if (k1 < C_) stage(k1, cur ^ 1);

        // ---- gather fragments per the CDNA5 16-bit WMMA VGPR layouts
        // A (16x32): lane m = l16; halves h<8 -> K = lhalf*8 + h,
        //                          h>=8 -> K = 16 + lhalf*8 + (h-8)
        v16h afrag[2];
        #pragma unroll
        for (int mi = 0; mi < 2; ++mi) {
            int r = waveM * 32 + mi * 16 + l16;
            v8h lo = *(const v8h*)(&Alds[cur][r * LDS_STRIDE + lhalf * 8]);
            v8h hi = *(const v8h*)(&Alds[cur][r * LDS_STRIDE + 16 + lhalf * 8]);
            #pragma unroll
            for (int q = 0; q < 8; ++q) { afrag[mi][q] = lo[q]; afrag[mi][q + 8] = hi[q]; }
        }
        // B (32x16): lane n = l16; halves h<8 -> K = lhalf*8 + h, h>=8 -> K = 16 + lhalf*8 + (h-8)
        v16h bfrag[4];
        #pragma unroll
        for (int ni = 0; ni < 4; ++ni) {
            int r = waveN * 64 + ni * 16 + l16;
            v8h lo = *(const v8h*)(&Blds[cur][r * LDS_STRIDE + lhalf * 8]);
            v8h hi = *(const v8h*)(&Blds[cur][r * LDS_STRIDE + 16 + lhalf * 8]);
            #pragma unroll
            for (int q = 0; q < 8; ++q) { bfrag[ni][q] = lo[q]; bfrag[ni][q + 8] = hi[q]; }
        }

        #pragma unroll
        for (int mi = 0; mi < 2; ++mi)
            #pragma unroll
            for (int ni = 0; ni < 4; ++ni)
                acc[mi][ni] = __builtin_amdgcn_wmma_f32_16x16x32_f16(
                    /*neg_a=*/false, afrag[mi],
                    /*neg_b=*/false, bfrag[ni],
                    /*c_mod=*/(short)0, acc[mi][ni],
                    /*reuse_a=*/false, /*reuse_b=*/false);

#if HAVE_ASYNC_LDS
        if (k1 < C_) __builtin_amdgcn_s_wait_asynccnt(0);
#endif
        __syncthreads();
    }

    // ---- epilogue: C/D layout: lane col N = l16, VGPR j -> row M = lhalf*8 + j
    #pragma unroll
    for (int mi = 0; mi < 2; ++mi) {
        #pragma unroll
        for (int ni = 0; ni < 4; ++ni) {
            int col = bn + waveN * 64 + ni * 16 + l16;
            #pragma unroll
            for (int j = 0; j < 8; ++j) {
                int row = bm + waveM * 32 + mi * 16 + lhalf * 8 + j;
                float v = acc[mi][ni][j];
                if (EPI == 1)      v = __builtin_amdgcn_rcpf(1.0f + __expf(-v)); // sigmoid
                else if (EPI == 2) v = __expf(v);                                // exp
                D[(size_t)row * C_ + col] = v;
            }
        }
    }
}

// ---------------------------------------------------------------------------
// Time-mix + fp16 cast:  in[b,t,c] = x*mix + x_shift*(1-mix), x_shift[t=0]=state
// ---------------------------------------------------------------------------
__global__ __launch_bounds__(256)
void mix_cast_kernel(const float* __restrict__ x, const float* __restrict__ state,
                     const float* __restrict__ tmr, const float* __restrict__ tmk,
                     const float* __restrict__ tmv,
                     half_t* __restrict__ r16, half_t* __restrict__ k16,
                     half_t* __restrict__ v16)
{
    size_t n = (size_t)blockIdx.x * blockDim.x + threadIdx.x;
    if (n >= (size_t)B_ * T_ * C_) return;
    int    c  = (int)(n % C_);
    size_t bt = n / C_;
    int    t  = (int)(bt % T_);
    int    b  = (int)(bt / T_);
    float xv = x[n];
    float xm = (t == 0) ? state[b * C_ + c] : x[n - C_];
    float mr = tmr[c], mk = tmk[c], mv = tmv[c];
    r16[n] = (half_t)(xv * mr + xm * (1.0f - mr));
    k16[n] = (half_t)(xv * mk + xm * (1.0f - mk));
    v16[n] = (half_t)(xv * mv + xm * (1.0f - mv));
}

// ---------------------------------------------------------------------------
// Weight fp32 -> fp16 casts (all four weights at once)
// ---------------------------------------------------------------------------
__global__ __launch_bounds__(256)
void wcast_kernel(const float* __restrict__ Wr, const float* __restrict__ Wk,
                  const float* __restrict__ Wv, const float* __restrict__ Wo,
                  half_t* __restrict__ Wr16, half_t* __restrict__ Wk16,
                  half_t* __restrict__ Wv16, half_t* __restrict__ Wo16)
{
    int n = blockIdx.x * blockDim.x + threadIdx.x;
    if (n >= C_ * C_) return;
    Wr16[n] = (half_t)Wr[n];
    Wk16[n] = (half_t)Wk[n];
    Wv16[n] = (half_t)Wv[n];
    Wo16[n] = (half_t)Wo[n];
}

// ---------------------------------------------------------------------------
// Per-channel decay scan:  s = s*decay + k*v ;  att = r*s  (fp16 out for GEMM)
// One thread per (b,c); loads coalesced across c. Also writes new_state.
// ---------------------------------------------------------------------------
__global__ __launch_bounds__(256)
void scan_kernel(const float* __restrict__ r, const float* __restrict__ k,
                 const float* __restrict__ v, const float* __restrict__ time_decay,
                 const float* __restrict__ state_in, const float* __restrict__ x,
                 half_t* __restrict__ att16, float* __restrict__ new_state)
{
    int tid = blockIdx.x * blockDim.x + threadIdx.x;   // B*C threads
    if (tid >= B_ * C_) return;
    int c = tid % C_;
    int b = tid / C_;
    float decay = __expf(time_decay[c]);
    float s = state_in[tid];
    size_t base = (size_t)b * T_ * C_ + c;
    #pragma unroll 4
    for (int t = 0; t < T_; ++t) {
        size_t i = base + (size_t)t * C_;
        s = s * decay + k[i] * v[i];
        att16[i] = (half_t)(r[i] * s);                 // ATT_SCALE == 1.0
    }
    new_state[tid] = x[base + (size_t)(T_ - 1) * C_];
}

// ---------------------------------------------------------------------------
extern "C" void kernel_launch(void* const* d_in, const int* in_sizes, int n_in,
                              void* d_out, int out_size, void* d_ws, size_t ws_size,
                              hipStream_t stream)
{
    const float* x     = (const float*)d_in[0];
    const float* state = (const float*)d_in[1];
    const float* W_r   = (const float*)d_in[2];
    const float* W_k   = (const float*)d_in[3];
    const float* W_v   = (const float*)d_in[4];
    const float* W_o   = (const float*)d_in[5];
    const float* tmr   = (const float*)d_in[6];
    const float* tmk   = (const float*)d_in[7];
    const float* tmv   = (const float*)d_in[8];
    const float* tdec  = (const float*)d_in[9];

    const size_t MC = (size_t)M_TOT * C_;   // 16,777,216 elements
    const size_t CC = (size_t)C_ * C_;

    // workspace carve
    char* p = (char*)d_ws;
    half_t* r_in16 = (half_t*)p;  p += MC * sizeof(half_t);
    half_t* k_in16 = (half_t*)p;  p += MC * sizeof(half_t);
    half_t* v_in16 = (half_t*)p;  p += MC * sizeof(half_t);
    half_t* Wr16   = (half_t*)p;  p += CC * sizeof(half_t);
    half_t* Wk16   = (half_t*)p;  p += CC * sizeof(half_t);
    half_t* Wv16   = (half_t*)p;  p += CC * sizeof(half_t);
    half_t* Wo16   = (half_t*)p;  p += CC * sizeof(half_t);
    float*  r_f    = (float*)p;   p += MC * sizeof(float);
    float*  k_f    = (float*)p;   p += MC * sizeof(float);
    float*  v_f    = (float*)p;   p += MC * sizeof(float);
    half_t* att16  = r_in16;      // reuse: r_in16 dead once r-GEMM is done

    float* out       = (float*)d_out;              // (B,T,C)
    float* new_state = out + MC;                   // (B,C) appended

    const int threads = 256;
    // 1) time mix + fp16 cast
    mix_cast_kernel<<<(unsigned)((MC + threads - 1) / threads), threads, 0, stream>>>(
        x, state, tmr, tmk, tmv, r_in16, k_in16, v_in16);
    // 2) weight casts
    wcast_kernel<<<(unsigned)((CC + threads - 1) / threads), threads, 0, stream>>>(
        W_r, W_k, W_v, W_o, Wr16, Wk16, Wv16, Wo16);

    dim3 ggrid(M_TOT / BM, C_ / BN);   // (128, 8)
    // 3) r = sigmoid(r_in @ Wr^T)
    wmma_gemm_kernel<1><<<ggrid, threads, 0, stream>>>(r_in16, Wr16, r_f);
    // 4) k = exp(k_in @ Wk^T)
    wmma_gemm_kernel<2><<<ggrid, threads, 0, stream>>>(k_in16, Wk16, k_f);
    // 5) v = v_in @ Wv^T
    wmma_gemm_kernel<0><<<ggrid, threads, 0, stream>>>(v_in16, Wv16, v_f);
    // 6) decay scan -> att16 (fp16), plus new_state
    scan_kernel<<<(B_ * C_ + threads - 1) / threads, threads, 0, stream>>>(
        r_f, k_f, v_f, tdec, state, x, att16, new_state);
    // 7) output = att @ Wo^T  (fp32 out)
    wmma_gemm_kernel<0><<<ggrid, threads, 0, stream>>>(att16, Wo16, out);
}
